// RnCLoss_33243046871823
// MI455X (gfx1250) — matrix operand order repaired and checked by reading, not compiled, not run on hardware
//
#include <hip/hip_runtime.h>
#include <hip/hip_bf16.h>
#include <math.h>

// CDNA5 WMMA operand types (probe-confirmed signatures)
typedef __attribute__((ext_vector_type(16))) __bf16 v16bf;
typedef __attribute__((ext_vector_type(8)))  float  v8f;

#define N_TOT 512   // 2*B
#define DIM   256   // feature dim
#define INV_TEMP 0.5f
#define EPS_  1e-8f

static __device__ __forceinline__ unsigned short f2bf_rne(float x) {
    unsigned int u = __float_as_uint(x);
    unsigned int r = u + 0x7FFFu + ((u >> 16) & 1u);   // round-to-nearest-even
    return (unsigned short)(r >> 16);
}
static __device__ __forceinline__ float bf2f(unsigned short h) {
    return __uint_as_float(((unsigned int)h) << 16);
}

// ---------------------------------------------------------------------------
// Kernel 1: gather f = concat(features[:,0], features[:,1]) and split each
// fp32 into bf16 hi/lo halves for the bf16x3 WMMA Gram matrix.
// ---------------------------------------------------------------------------
__global__ void __launch_bounds__(DIM)
rnc_split_bf16(const float* __restrict__ features,
               unsigned short* __restrict__ fhi,
               unsigned short* __restrict__ flo) {
    const int i = blockIdx.x;    // logical row 0..511
    const int d = threadIdx.x;   // 0..255
    // features is [256, 2, 256]: row stride 512, view-select stride 256
    const float* src = features + ((i < 256) ? (size_t)i * 512
                                             : (size_t)(i - 256) * 512 + 256);
    float x = src[d];
    unsigned short h = f2bf_rne(x);
    unsigned short l = f2bf_rne(x - bf2f(h));
    fhi[(size_t)i * DIM + d] = h;
    flo[(size_t)i * DIM + d] = l;
}

// ---------------------------------------------------------------------------
// Kernel 2: G = F F^T via v_wmma_f32_16x16x32_bf16, bf16x3 split
// (hi*hi + hi*lo + lo*hi) to recover ~fp32 accuracy.
// One wave per 16x16 output tile; 4 waves per block; 32x32 tiles total.
// K = 256 -> 8 chunks of 32 -> 24 WMMA per tile.
// ---------------------------------------------------------------------------
__global__ void __launch_bounds__(128)
rnc_gram_wmma(const unsigned short* __restrict__ fhi,
              const unsigned short* __restrict__ flo,
              float* __restrict__ G) {
    const int lane = threadIdx.x;                       // 0..31 (wave32)
    const int tile = blockIdx.x * 4 + threadIdx.y;      // 0..1023
    const int ti   = (tile >> 5) * 16;                  // output row base
    const int tj   = (tile & 31) * 16;                  // output col base

    const int half = lane >> 4;                         // lane group 0/1
    const int mn   = lane & 15;                         // row (A) / col (B)

    union { v16bf v; unsigned int u[8]; } ahi, alo, bhi, blo;
    v8f acc = {};

    const unsigned int* arow_h = (const unsigned int*)(fhi + (size_t)(ti + mn) * DIM);
    const unsigned int* arow_l = (const unsigned int*)(flo + (size_t)(ti + mn) * DIM);
    const unsigned int* brow_h = (const unsigned int*)(fhi + (size_t)(tj + mn) * DIM);
    const unsigned int* brow_l = (const unsigned int*)(flo + (size_t)(tj + mn) * DIM);

    for (int kb = 0; kb < DIM; kb += 32) {
        #pragma unroll
        for (int t = 0; t < 8; ++t) {
            // A fragment (16x32): lanes 0-15 hold K {0..7,16..23}, lanes 16-31 K {8..15,24..31}
            int ka  = kb + ((t < 4) ? 0 : 16) + half * 8 + 2 * (t & 3);
            // B fragment (32x16): lanes 0-15 hold K {0..7,16..23}, lanes 16-31 K {8..15,24..31}
            int kbk = kb + ((t < 4) ? 0 : 16) + half * 8 + 2 * (t & 3);
            ahi.u[t] = arow_h[ka  >> 1];
            alo.u[t] = arow_l[ka  >> 1];
            bhi.u[t] = brow_h[kbk >> 1];
            blo.u[t] = brow_l[kbk >> 1];
        }
        acc = __builtin_amdgcn_wmma_f32_16x16x32_bf16(false, ahi.v, false, bhi.v,
                                                      (short)0, acc, false, false);
        acc = __builtin_amdgcn_wmma_f32_16x16x32_bf16(false, ahi.v, false, blo.v,
                                                      (short)0, acc, false, false);
        acc = __builtin_amdgcn_wmma_f32_16x16x32_bf16(false, alo.v, false, bhi.v,
                                                      (short)0, acc, false, false);
    }

    // C/D layout: VGPR r, lane l -> M = r + 8*(l>=16), N = l&15
    #pragma unroll
    for (int r = 0; r < 8; ++r) {
        G[(size_t)(ti + r + 8 * half) * N_TOT + tj + mn] = acc[r];
    }
}

// ---------------------------------------------------------------------------
// Kernel 3: per-row loss. dist2[i,j] = relu(G_ii + G_jj - 2 G_ij) is exactly
// zero on the diagonal (same-source norms), so the reference's row-max is
// identically 0 and exp(logits - max) == exp(logits). One block per row,
// columns staged in LDS, each thread owns one positive k and scans all j.
// ---------------------------------------------------------------------------
__global__ void __launch_bounds__(N_TOT)
rnc_row_loss(const float* __restrict__ G,
             const float* __restrict__ labels,
             float* __restrict__ partials) {
    __shared__ float sh_l[N_TOT];   // logits
    __shared__ float sh_e[N_TOT];   // exp(logits), 0 on diagonal
    __shared__ float sh_d[N_TOT];   // |L_i - L_j|
    __shared__ float sh_r[N_TOT];   // reduction

    const int i = blockIdx.x;
    const int j = threadIdx.x;

    float Gii = G[(size_t)i * N_TOT + i];
    float Gjj = G[(size_t)j * N_TOT + j];
    float Gij = G[(size_t)i * N_TOT + j];
    float d2  = fmaxf(Gii + Gjj - 2.0f * Gij, 0.0f);
    float lg  = -sqrtf(d2) * INV_TEMP;
    float Li  = labels[i & 255];
    float Lj  = labels[j & 255];

    sh_l[j] = lg;
    sh_e[j] = (j == i) ? 0.0f : expf(lg);   // drop diagonal column
    sh_d[j] = fabsf(Li - Lj);
    __syncthreads();

    float part = 0.0f;
    if (j != i) {                            // positive k = j
        const float dk = sh_d[j];
        float denom = 0.0f;
        #pragma unroll 8
        for (int t = 0; t < N_TOT; ++t)
            denom += (sh_d[t] >= dk) ? sh_e[t] : 0.0f;  // t==i contributes 0
        part = logf(denom + EPS_) - sh_l[j];
    }
    sh_r[j] = part;
    __syncthreads();
    for (int s = N_TOT / 2; s > 0; s >>= 1) {
        if (j < s) sh_r[j] += sh_r[j + s];
        __syncthreads();
    }
    if (j == 0) partials[i] = sh_r[0];
}

// ---------------------------------------------------------------------------
// Kernel 4: deterministic final reduction (no fp atomics).
// ---------------------------------------------------------------------------
__global__ void __launch_bounds__(N_TOT)
rnc_final(const float* __restrict__ partials, float* __restrict__ out) {
    __shared__ float sh[N_TOT];
    const int j = threadIdx.x;
    sh[j] = partials[j];
    __syncthreads();
    for (int s = N_TOT / 2; s > 0; s >>= 1) {
        if (j < s) sh[j] += sh[j + s];
        __syncthreads();
    }
    if (j == 0) out[0] = sh[0] / ((float)N_TOT * (float)(N_TOT - 1));
}

extern "C" void kernel_launch(void* const* d_in, const int* in_sizes, int n_in,
                              void* d_out, int out_size, void* d_ws, size_t ws_size,
                              hipStream_t stream) {
    (void)in_sizes; (void)n_in; (void)out_size; (void)ws_size;
    const float* features = (const float*)d_in[0];   // [256,2,256] fp32
    const float* labels   = (const float*)d_in[1];   // [256] fp32

    char* ws = (char*)d_ws;
    float*          G    = (float*)ws;                                        // 1 MB
    unsigned short* fhi  = (unsigned short*)(ws + (size_t)N_TOT * N_TOT * 4); // 256 KB
    unsigned short* flo  = fhi + (size_t)N_TOT * DIM;                         // 256 KB
    float*          part = (float*)(ws + (size_t)N_TOT * N_TOT * 4
                                       + 2 * (size_t)N_TOT * DIM * 2);        // 2 KB

    rnc_split_bf16<<<N_TOT, DIM, 0, stream>>>(features, fhi, flo);
    rnc_gram_wmma <<<dim3(1024 / 4), dim3(32, 4), 0, stream>>>(fhi, flo, G);
    rnc_row_loss  <<<N_TOT, N_TOT, 0, stream>>>(G, labels, part);
    rnc_final     <<<1, N_TOT, 0, stream>>>(part, (float*)d_out);
}
